// Seq2Seq_64699387347255
// MI455X (gfx1250) — matrix operand order repaired and checked by reading, not compiled, not run on hardware
//
#include <hip/hip_runtime.h>
#include <math.h>

// ---------------------------------------------------------------------------
// Seq2Seq (tanh RNN enc/dec + vocab projection + log_softmax) for gfx1250.
// Strategy:
//   * everything matrix-shaped runs through v_wmma_f32_16x16x32_f16
//   * x@Wih + (bih+bhh) hoisted out of the recurrence into one big GEMM
//   * recurrence: persistent 8-WG kernel, atomic grid barrier per step
//   * projection GEMM streams f16 weights from the 192MB L2
//   * log-softmax: online row LSE + normalize pass
// ---------------------------------------------------------------------------

typedef __attribute__((ext_vector_type(16))) _Float16 v16h;
typedef __attribute__((ext_vector_type(8)))  _Float16 v8h;
typedef __attribute__((ext_vector_type(8)))  float    v8f;

#define B_   32
#define LS_  64
#define LT_  64
#define V_   32000
#define E_   512
#define H_   512
#define NROW 2048          // B * L
#define SCAN_WGS 8         // persistent workgroups in the recurrence kernel

// ---------------- WMMA fragment loaders (contiguous k order, same in A and B) -----

// A: 16x32 f16 tile, row-major source with leading dim lda.
// lane m = lane&15 holds row M=m; halves[0..15]=K(hi*16 ..)
__device__ __forceinline__ v16h load_a_frag(const _Float16* __restrict__ A,
                                            int lda, int row0, int k0) {
  const int lane = threadIdx.x & 31;
  const int m = lane & 15, hi = lane >> 4;
  const _Float16* p = A + (size_t)(row0 + m) * lda + k0 + hi * 16;
  v8h lo = *(const v8h*)(p);
  v8h hv = *(const v8h*)(p + 8);
  v16h r;
#pragma unroll
  for (int i = 0; i < 8; ++i) { r[i] = lo[i]; r[i + 8] = hv[i]; }
  return r;
}

// B: 32x16 f16 tile taken from a pre-TRANSPOSED weight (BT is N x K row-major).
// lane n = lane&15 holds column N=n; halves[j] = B[k0 + hi*16 + j][n]
//   = BT[col0+n][k0 + hi*16 + j]  -> one contiguous 32-byte load per lane.
__device__ __forceinline__ v16h load_b_frag(const _Float16* __restrict__ BT,
                                            int ldb, int col0, int k0) {
  const int lane = threadIdx.x & 31;
  const int n = lane & 15, hi = lane >> 4;
  const _Float16* p = BT + (size_t)(col0 + n) * ldb + k0 + hi * 16;
  return *(const v16h*)(p);
}

// D/C 16x16 f32: VGPR r -> row = r + (lane>=16)*8, col = lane&15

// ---------------- small prep kernels ---------------------------------------

// f32 (R x C) -> f16 transposed (C x R), LDS-tiled so stores stay coalesced.
__global__ __launch_bounds__(256) void conv_t_tiled(const float* __restrict__ in,
                                                    _Float16* __restrict__ outT,
                                                    int R, int C) {
  __shared__ _Float16 tile[32][33];
  const int tc = blockIdx.x * 32, tr = blockIdx.y * 32;
#pragma unroll
  for (int i = threadIdx.y; i < 32; i += 8) {
    int r = tr + i, c = tc + threadIdx.x;
    tile[i][threadIdx.x] = (r < R && c < C) ? (_Float16)in[(size_t)r * C + c]
                                            : (_Float16)0.f;
  }
  __syncthreads();
#pragma unroll
  for (int i = threadIdx.y; i < 32; i += 8) {
    int c = tc + i, r = tr + threadIdx.x;
    if (c < C && r < R) outT[(size_t)c * R + r] = tile[threadIdx.x][i];
  }
}

__global__ __launch_bounds__(256) void bias_sum_kernel(const float* __restrict__ a,
                                                       const float* __restrict__ b,
                                                       float* __restrict__ o, int n) {
  int i = blockIdx.x * 256 + threadIdx.x;
  if (i < n) o[i] = a[i] + b[i];
}

// tokens (B x L) -> X f16 (NROW x E), row index = t*B + b (time-major rows)
__global__ __launch_bounds__(256) void gather_embed(const int* __restrict__ tokens,
                                                    const float* __restrict__ embed,
                                                    _Float16* __restrict__ X, int L) {
  const int row = blockIdx.x;            // 0..NROW-1
  const int t = row / B_, b = row % B_;
  const int tok = tokens[b * L + t];
  const float* src = embed + (size_t)tok * E_;
  _Float16* dst = X + (size_t)row * E_;
  for (int i = threadIdx.x; i < E_; i += 256) dst[i] = (_Float16)src[i];
}

// ---------------- generic WMMA GEMM: C(MxN,f32) = A(MxK,f16) * BT(NxK,f16) + bias[n]
// workgroup = 256 thr (8 waves), covers 64(M) x 128(N); wave tile 32x32 (2x2 frags)
__global__ __launch_bounds__(256) void wmma_gemm_bias(const _Float16* __restrict__ A,
                                                      const _Float16* __restrict__ BT,
                                                      const float* __restrict__ bias,
                                                      float* __restrict__ C,
                                                      int N, int K) {
  const int wave = threadIdx.x >> 5, lane = threadIdx.x & 31;
  const int hi = lane >> 4, nn = lane & 15;
  const int row0 = blockIdx.y * 64 + (wave & 1) * 32;
  const int col0 = blockIdx.x * 128 + (wave >> 1) * 32;
  v8f acc[2][2] = {};
  for (int k0 = 0; k0 < K; k0 += 32) {
    v16h a0 = load_a_frag(A, K, row0,      k0);
    v16h a1 = load_a_frag(A, K, row0 + 16, k0);
    v16h b0 = load_b_frag(BT, K, col0,      k0);
    v16h b1 = load_b_frag(BT, K, col0 + 16, k0);
    acc[0][0] = __builtin_amdgcn_wmma_f32_16x16x32_f16(false, a0, false, b0, (short)0, acc[0][0], false, false);
    acc[0][1] = __builtin_amdgcn_wmma_f32_16x16x32_f16(false, a0, false, b1, (short)0, acc[0][1], false, false);
    acc[1][0] = __builtin_amdgcn_wmma_f32_16x16x32_f16(false, a1, false, b0, (short)0, acc[1][0], false, false);
    acc[1][1] = __builtin_amdgcn_wmma_f32_16x16x32_f16(false, a1, false, b1, (short)0, acc[1][1], false, false);
  }
#pragma unroll
  for (int mi = 0; mi < 2; ++mi)
#pragma unroll
    for (int ni = 0; ni < 2; ++ni)
#pragma unroll
      for (int r = 0; r < 8; ++r) {
        int row = row0 + mi * 16 + hi * 8 + r;
        int col = col0 + ni * 16 + nn;
        C[(size_t)row * N + col] = acc[mi][ni][r] + bias[col];
      }
}

// ---------------- persistent recurrence kernel -----------------------------
// h_{t+1} = tanh( xW[t] + h_t @ Whh )  over 64 enc + 64 dec steps.
// 8 WGs x 8 waves = 64 waves = the 2x32 16x16 tiles of the 32x512 hidden state.
__global__ __launch_bounds__(256) void rnn_scan(const float* __restrict__ xW_enc,
                                                const float* __restrict__ xW_dec,
                                                const _Float16* __restrict__ WhhT_enc,
                                                const _Float16* __restrict__ WhhT_dec,
                                                _Float16* __restrict__ h0buf,
                                                _Float16* __restrict__ h1buf,
                                                _Float16* __restrict__ dechs,
                                                unsigned* __restrict__ counter) {
  const int wgl = blockIdx.x * 8 + (threadIdx.x >> 5);   // 0..63 global wave id
  const int lane = threadIdx.x & 31;
  const int hi = lane >> 4, nn = lane & 15;
  const int row0 = (wgl & 1) * 16;        // over B=32
  const int col0 = (wgl >> 1) * 16;       // over H=512

  for (int step = 0; step < LS_ + LT_; ++step) {
    const bool dec = (step >= LS_);
    const int t = dec ? step - LS_ : step;
    const float*    xW = dec ? xW_dec   : xW_enc;
    const _Float16* WT = dec ? WhhT_dec : WhhT_enc;
    const _Float16* hcur = (step & 1) ? h1buf : h0buf;
    _Float16*       hnxt = (step & 1) ? h0buf : h1buf;

    // accumulator initialized with the hoisted xW[t] + bias tile
    v8f acc;
#pragma unroll
    for (int r = 0; r < 8; ++r) {
      int row = row0 + hi * 8 + r;
      acc[r] = xW[((size_t)t * B_ + row) * H_ + col0 + nn];
    }
#pragma unroll 4
    for (int k0 = 0; k0 < H_; k0 += 32) {
      v16h a = load_a_frag(hcur, H_, row0, k0);
      v16h b = load_b_frag(WT,   H_, col0, k0);
      acc = __builtin_amdgcn_wmma_f32_16x16x32_f16(false, a, false, b, (short)0, acc, false, false);
    }
#pragma unroll
    for (int r = 0; r < 8; ++r) {
      int row = row0 + hi * 8 + r;
      _Float16 hv = (_Float16)tanhf(acc[r]);
      hnxt[(size_t)row * H_ + col0 + nn] = hv;
      if (dec) dechs[((size_t)t * B_ + row) * H_ + col0 + nn] = hv;
    }

    // device-scope grid barrier between steps (monotonic counter)
    __threadfence();
    __syncthreads();
    if (threadIdx.x == 0) {
      atomicAdd(counter, 1u);
      const unsigned target = (unsigned)SCAN_WGS * (step + 1);
      while (*(volatile unsigned*)counter < target) __builtin_amdgcn_s_sleep(1);
    }
    __syncthreads();
    __threadfence();   // acquire: invalidate near caches before reading hnxt
  }
}

// ---------------- projection GEMM: logits -> d_out (B, Lt, V) --------------
__global__ __launch_bounds__(256) void proj_gemm(const _Float16* __restrict__ A,
                                                 const _Float16* __restrict__ BT,
                                                 const float* __restrict__ bias,
                                                 float* __restrict__ out) {
  const int wave = threadIdx.x >> 5, lane = threadIdx.x & 31;
  const int hi = lane >> 4, nn = lane & 15;
  const int row0 = blockIdx.y * 64 + (wave & 1) * 32;
  const int col0 = blockIdx.x * 128 + (wave >> 1) * 32;
  v8f acc[2][2] = {};
  for (int k0 = 0; k0 < H_; k0 += 32) {
    v16h a0 = load_a_frag(A, H_, row0,      k0);
    v16h a1 = load_a_frag(A, H_, row0 + 16, k0);
    v16h b0 = load_b_frag(BT, H_, col0,      k0);
    v16h b1 = load_b_frag(BT, H_, col0 + 16, k0);
    acc[0][0] = __builtin_amdgcn_wmma_f32_16x16x32_f16(false, a0, false, b0, (short)0, acc[0][0], false, false);
    acc[0][1] = __builtin_amdgcn_wmma_f32_16x16x32_f16(false, a0, false, b1, (short)0, acc[0][1], false, false);
    acc[1][0] = __builtin_amdgcn_wmma_f32_16x16x32_f16(false, a1, false, b0, (short)0, acc[1][0], false, false);
    acc[1][1] = __builtin_amdgcn_wmma_f32_16x16x32_f16(false, a1, false, b1, (short)0, acc[1][1], false, false);
  }
#pragma unroll
  for (int mi = 0; mi < 2; ++mi)
#pragma unroll
    for (int ni = 0; ni < 2; ++ni)
#pragma unroll
      for (int r = 0; r < 8; ++r) {
        int row = row0 + mi * 16 + hi * 8 + r;   // row = t*B + b
        int col = col0 + ni * 16 + nn;
        int t = row / B_, b = row % B_;
        out[((size_t)(b * LT_ + t)) * V_ + col] = acc[mi][ni][r] + bias[col];
      }
}

// ---------------- log-softmax: per-row online LSE, then subtract -----------
__global__ __launch_bounds__(256) void row_lse(const float* __restrict__ out,
                                               float* __restrict__ lse) {
  const float* p = out + (size_t)blockIdx.x * V_;
  float m = -INFINITY, s = 0.f;
  for (int i = threadIdx.x; i < V_; i += 256) {
    float x = p[i];
    if (x > m) { s = s * __expf(m - x) + 1.f; m = x; }
    else       { s += __expf(x - m); }
  }
  __shared__ float sm[256], ss[256];
  sm[threadIdx.x] = m; ss[threadIdx.x] = s;
  __syncthreads();
  for (int off = 128; off > 0; off >>= 1) {
    if ((int)threadIdx.x < off) {
      float m2 = sm[threadIdx.x + off], s2 = ss[threadIdx.x + off];
      float M = fmaxf(sm[threadIdx.x], m2);
      ss[threadIdx.x] = ss[threadIdx.x] * __expf(sm[threadIdx.x] - M) + s2 * __expf(m2 - M);
      sm[threadIdx.x] = M;
    }
    __syncthreads();
  }
  if (threadIdx.x == 0) lse[blockIdx.x] = sm[0] + __logf(ss[0]);
}

__global__ __launch_bounds__(256) void normalize_kernel(float* __restrict__ out,
                                                        const float* __restrict__ lse) {
  const size_t i4 = (size_t)blockIdx.x * 256 + threadIdx.x;
  if (i4 >= (size_t)NROW * (V_ / 4)) return;
  const float l = lse[i4 / (V_ / 4)];
  float4* p = ((float4*)out) + i4;
  float4 v = *p;
  v.x -= l; v.y -= l; v.z -= l; v.w -= l;
  *p = v;
}

// ---------------- host-side orchestration ----------------------------------

static inline size_t align256(size_t x) { return (x + 255) & ~(size_t)255; }

extern "C" void kernel_launch(void* const* d_in, const int* in_sizes, int n_in,
                              void* d_out, int out_size, void* d_ws, size_t ws_size,
                              hipStream_t stream) {
  (void)in_sizes; (void)n_in; (void)out_size;
  const int*   src       = (const int*)d_in[0];
  const int*   tgt       = (const int*)d_in[1];
  const float* enc_embed = (const float*)d_in[2];
  const float* enc_wih   = (const float*)d_in[3];
  const float* enc_bih   = (const float*)d_in[4];
  const float* enc_whh   = (const float*)d_in[5];
  const float* enc_bhh   = (const float*)d_in[6];
  const float* dec_embed = (const float*)d_in[7];
  const float* dec_wih   = (const float*)d_in[8];
  const float* dec_bih   = (const float*)d_in[9];
  const float* dec_whh   = (const float*)d_in[10];
  const float* dec_bhh   = (const float*)d_in[11];
  const float* h2o_w     = (const float*)d_in[12];
  const float* h2o_b     = (const float*)d_in[13];
  float* out = (float*)d_out;

  // workspace carve-up (~49 MB total)
  char* ws = (char*)d_ws;
  size_t off = 0;
  unsigned*  counter   = (unsigned*)(ws + off);  off = align256(off + 256);
  _Float16*  h0buf     = (_Float16*)(ws + off);  off = align256(off + (size_t)B_ * H_ * 2);
  _Float16*  h1buf     = (_Float16*)(ws + off);  off = align256(off + (size_t)B_ * H_ * 2);
  _Float16*  WihT_enc  = (_Float16*)(ws + off);  off = align256(off + (size_t)E_ * H_ * 2);
  _Float16*  WhhT_enc  = (_Float16*)(ws + off);  off = align256(off + (size_t)H_ * H_ * 2);
  _Float16*  WihT_dec  = (_Float16*)(ws + off);  off = align256(off + (size_t)E_ * H_ * 2);
  _Float16*  WhhT_dec  = (_Float16*)(ws + off);  off = align256(off + (size_t)H_ * H_ * 2);
  _Float16*  h2oT      = (_Float16*)(ws + off);  off = align256(off + (size_t)V_ * H_ * 2);
  _Float16*  Xsrc      = (_Float16*)(ws + off);  off = align256(off + (size_t)NROW * E_ * 2);
  _Float16*  Xtgt      = (_Float16*)(ws + off);  off = align256(off + (size_t)NROW * E_ * 2);
  float*     xW_enc    = (float*)(ws + off);     off = align256(off + (size_t)NROW * H_ * 4);
  float*     xW_dec    = (float*)(ws + off);     off = align256(off + (size_t)NROW * H_ * 4);
  _Float16*  dechs     = (_Float16*)(ws + off);  off = align256(off + (size_t)NROW * H_ * 4);
  float*     bias_enc  = (float*)(ws + off);     off = align256(off + (size_t)H_ * 4);
  float*     bias_dec  = (float*)(ws + off);     off = align256(off + (size_t)H_ * 4);
  float*     lse       = (float*)(ws + off);     off = align256(off + (size_t)NROW * 4);
  (void)ws_size;

  // reset barrier counter + h0 = 0 every call (graph-capture-safe memset node)
  hipMemsetAsync(counter, 0, 256 + (size_t)B_ * H_ * 2, stream);

  // weight convert+transpose to f16 (B-operand wants N x K row-major)
  conv_t_tiled<<<dim3(H_ / 32, E_ / 32), dim3(32, 8), 0, stream>>>(enc_wih, WihT_enc, E_, H_);
  conv_t_tiled<<<dim3(H_ / 32, H_ / 32), dim3(32, 8), 0, stream>>>(enc_whh, WhhT_enc, H_, H_);
  conv_t_tiled<<<dim3(H_ / 32, E_ / 32), dim3(32, 8), 0, stream>>>(dec_wih, WihT_dec, E_, H_);
  conv_t_tiled<<<dim3(H_ / 32, H_ / 32), dim3(32, 8), 0, stream>>>(dec_whh, WhhT_dec, H_, H_);
  conv_t_tiled<<<dim3(V_ / 32, H_ / 32), dim3(32, 8), 0, stream>>>(h2o_w, h2oT, H_, V_);

  bias_sum_kernel<<<2, 256, 0, stream>>>(enc_bih, enc_bhh, bias_enc, H_);
  bias_sum_kernel<<<2, 256, 0, stream>>>(dec_bih, dec_bhh, bias_dec, H_);

  // embedding gathers (time-major rows: t*B + b)
  gather_embed<<<NROW, 256, 0, stream>>>(src, enc_embed, Xsrc, LS_);
  gather_embed<<<NROW, 256, 0, stream>>>(tgt, dec_embed, Xtgt, LT_);

  // hoisted input projections: xW = X @ Wih + (bih + bhh)
  wmma_gemm_bias<<<dim3(H_ / 128, NROW / 64), 256, 0, stream>>>(Xsrc, WihT_enc, bias_enc, xW_enc, H_, E_);
  wmma_gemm_bias<<<dim3(H_ / 128, NROW / 64), 256, 0, stream>>>(Xtgt, WihT_dec, bias_dec, xW_dec, H_, E_);

  // 128-step recurrence in one persistent kernel
  rnn_scan<<<SCAN_WGS, 256, 0, stream>>>(xW_enc, xW_dec, WhhT_enc, WhhT_dec,
                                         h0buf, h1buf, dechs, counter);

  // vocab projection -> logits straight into d_out (B, Lt, V layout)
  proj_gemm<<<dim3(V_ / 128, NROW / 64), 256, 0, stream>>>(dechs, h2oT, h2o_b, out);

  // log-softmax over V
  row_lse<<<NROW, 256, 0, stream>>>(out, lse);
  normalize_kernel<<<(int)(((size_t)NROW * (V_ / 4) + 255) / 256), 256, 0, stream>>>(out, lse);
}
